// RG_LSTM_44478681317600
// MI455X (gfx1250) — matrix operand order repaired and checked by reading, not compiled, not run on hardware
//
#include <hip/hip_runtime.h>
#include <hip/hip_bf16.h>

// Problem constants (from the reference)
#define NN    2000   // nodes
#define NPAD  2048   // nodes padded for K-loop of the graph GEMM
#define TT    365    // timesteps
#define HH    128    // hidden
#define KX    64     // packed x-part K: 32 dyn + 16 stat + 16 zero pad
#define G4    512    // 4*H gate columns
#define OUTD  16     // output dim

typedef __bf16 bf16t;
typedef __attribute__((ext_vector_type(16))) __bf16 v16bf;
typedef __attribute__((ext_vector_type(8)))  __bf16 v8bf;
typedef __attribute__((ext_vector_type(8)))  float  v8f;

__device__ __forceinline__ v8f wmma_bf16(v16bf a, v16bf b, v8f c) {
  // D = A(16x32 bf16) x B(32x16 bf16) + C(16x16 f32)
  return __builtin_amdgcn_wmma_f32_16x16x32_bf16(
      /*neg_a=*/false, a, /*neg_b=*/false, b,
      /*c_mod=*/(short)0, c, /*reuse_a=*/false, /*reuse_b=*/false);
}

// ---- Branchless activations (v_exp_f32 / v_rcp_f32; no EXEC churn) ----
__device__ __forceinline__ float fexp2_(float x) {
#if defined(__AMDGCN__) && __has_builtin(__builtin_amdgcn_exp2f)
  return __builtin_amdgcn_exp2f(x);
#else
  return exp2f(x);
#endif
}
__device__ __forceinline__ float frcp_(float x) {
#if defined(__AMDGCN__) && __has_builtin(__builtin_amdgcn_rcpf)
  return __builtin_amdgcn_rcpf(x);
#else
  return 1.0f / x;
#endif
}
__device__ __forceinline__ float fsig_(float x) {
  return frcp_(1.0f + fexp2_(x * -1.44269504088896340736f));
}
__device__ __forceinline__ float ftanh_(float x) {
  float xc = fminf(fmaxf(x, -15.0f), 15.0f);
  float e  = fexp2_(xc * -2.88539008177792681472f);  // exp(-2x)
  return (1.0f - e) * frcp_(1.0f + e);
}

// ---- Fragment loaders, layouts per CDNA5 ISA 7.12.2 (wave32) ----

// A-matrix 16x32 bf16 from row-major src (row stride ld, elements).
// lane<16: K = k0 + {0..7, 16..23}; lane>=16: K = k0 + {8..15, 24..31}; M = lane&15.
__device__ __forceinline__ v16bf load_a_bf16(const bf16t* base, int ld, int row0,
                                             int k0, int lane) {
  int m    = lane & 15;
  int koff = (lane >> 4) << 3;                 // 0 or 8
  const bf16t* p = base + (size_t)(row0 + m) * ld + k0 + koff;
  v8bf lo = *(const v8bf*)p;                   // K = k0+koff .. +7
  v8bf hi = *(const v8bf*)(p + 16);            // K = k0+koff+16 .. +23
  return __builtin_shufflevector(lo, hi, 0,1,2,3,4,5,6,7,8,9,10,11,12,13,14,15);
}

// B-matrix 32x16 bf16 from BT storage: bt[n][k] = B[k][n], row stride ld.
// Per lane: N = lane&15; lane<16: K = k0 + {0..7, 16..23}; lane>=16: K = k0 + {8..15, 24..31}.
__device__ __forceinline__ v16bf load_b_bf16(const bf16t* bt, int ld, int n0,
                                             int k0, int lane) {
  int n    = lane & 15;
  int koff = (lane >> 4) << 3;                 // 0 or 8
  const bf16t* p = bt + (size_t)(n0 + n) * ld + k0 + koff;
  v8bf lo = *(const v8bf*)p;
  v8bf hi = *(const v8bf*)(p + 16);
  return __builtin_shufflevector(lo, hi, 0,1,2,3,4,5,6,7,8,9,10,11,12,13,14,15);
}

// ---------------- One-time setup kernels ----------------

// AtBf[i][j] = bf16(A[j][i]) (zero for j>=NN); zero qT pad columns j>=NN.
__global__ void k_setup_A(const float* __restrict__ A, bf16t* __restrict__ AtBf,
                          bf16t* __restrict__ qT) {
  unsigned idx = blockIdx.x * 256u + threadIdx.x;
  if (idx < (unsigned)NN * NPAD) {
    int i = idx / NPAD, j = idx % NPAD;
    AtBf[idx] = (j < NN) ? (bf16t)A[(size_t)j * NN + i] : (bf16t)0.0f;
  }
  if (idx < (unsigned)HH * NPAD) {
    int j = idx % NPAD;
    if (j >= NN) qT[idx] = (bf16t)0.0f;
  }
}

// Weight conversion + zero-init of h/c state (f32) and bf16 h mirror.
__global__ void k_setup_W(const float* __restrict__ W_ih, const float* __restrict__ W_hh,
                          const float* __restrict__ W_q, bf16t* __restrict__ WxT,
                          bf16t* __restrict__ WhT, bf16t* __restrict__ WqBf,
                          float* __restrict__ h, float* __restrict__ c,
                          bf16t* __restrict__ hbf) {
  unsigned idx = blockIdx.x * 256u + threadIdx.x;
  if (idx < (unsigned)G4 * KX) {                     // [512][64]: W_ih | zero pad
    int g = idx / KX, k = idx % KX;
    WxT[idx] = (bf16t)((k < 48) ? W_ih[(size_t)g * 48 + k] : 0.0f);
  }
  if (idx < (unsigned)G4 * HH) WhT[idx] = (bf16t)W_hh[idx];   // [512][128]
  if (idx < (unsigned)HH * HH) WqBf[idx] = (bf16t)W_q[idx];   // [128][128]
  if (idx < (unsigned)NN * HH) { h[idx] = 0.0f; c[idx] = 0.0f; hbf[idx] = (bf16t)0.0f; }
}

// ---------------- Per-step kernels ----------------

// K1: q = tanh(h W_q^T + b_q) -> qT (bf16, transposed [HH][NPAD]).
// 8 waves/block, each one 16x16 tile; A-frags straight from bf16 h mirror.
__global__ __launch_bounds__(256) void k1_step(
    const bf16t* __restrict__ hbf, const bf16t* __restrict__ WqBf,
    const float* __restrict__ b_q, bf16t* __restrict__ qT) {
  int tid = threadIdx.x, lane = tid & 31, w = tid >> 5;
  int i0 = blockIdx.x * 16;
  int h0 = w * 16;
  int n = lane & 15, mbase = (lane >> 4) * 8;
  float bv = b_q[h0 + n];
  v8f acc;
#pragma unroll
  for (int v = 0; v < 8; ++v) acc[v] = bv;
#pragma unroll
  for (int k0 = 0; k0 < HH; k0 += 32) {
    v16bf a = load_a_bf16(hbf, HH, i0, k0, lane);
    v16bf b = load_b_bf16(WqBf, HH, h0, k0, lane);
    acc = wmma_bf16(a, b, acc);
  }
  v8bf qv;
#pragma unroll
  for (int v = 0; v < 8; ++v) qv[v] = (bf16t)ftanh_(acc[v]);
  *(v8bf*)(qT + (size_t)(h0 + n) * NPAD + i0 + mbase) = qv;  // transposed store
}

// K2: per block of 16 nodes; wave w owns hidden slice [w*16, w*16+16).
//   Phase A: c0 = c + At@q         (64 WMMAs, K=NPAD) -> registers
//   Phase B: i/f/g/o gate tiles     (24 WMMAs, shared A-frag x4) -> registers
//   Phase C: LSTM cell update fully in registers; writes c, h, h_bf16.
__global__ __launch_bounds__(256) void k2_step(
    int t, const float* __restrict__ x_d, const float* __restrict__ x_s,
    const bf16t* __restrict__ AtBf, const bf16t* __restrict__ qT,
    const bf16t* __restrict__ WxT, const bf16t* __restrict__ WhT,
    const float* __restrict__ b_ih, float* __restrict__ hstate,
    float* __restrict__ cstate, bf16t* hbf) {
  __shared__ __align__(16) bf16t xsh[16 * KX];  // 2 KB: [x_d(t) | x_s | 0] per node
  int tid = threadIdx.x, lane = tid & 31, w = tid >> 5;
  int i0 = blockIdx.x * 16;

  // Prefetch next step's dynamic inputs for this node tile (global_prefetch_b8).
  if (t + 1 < TT)
    __builtin_prefetch(x_d + ((size_t)(t + 1) * NN + i0) * 32, 0, 1);

  // Pack x tile into LDS (16 nodes x 64 bf16).
  {
    int m = tid >> 4, kl = tid & 15;
    const float* xdrow = x_d + ((size_t)t * NN + i0 + m) * 32;
    const float* xsrow = x_s + (size_t)(i0 + m) * 16;
#pragma unroll
    for (int r = 0; r < 4; ++r) {
      int k = kl + r * 16;
      float v = (k < 32) ? xdrow[k] : ((k < 48) ? xsrow[k - 32] : 0.0f);
      xsh[m * KX + k] = (bf16t)v;
    }
  }
  __syncthreads();

  int n = lane & 15, mbase = (lane >> 4) * 8;
  int h0 = w * 16;

  // ---- Phase A: graph propagation tile (registers), seeded with c ----
  v8f c0;
#pragma unroll
  for (int v = 0; v < 8; ++v)
    c0[v] = cstate[(size_t)(i0 + mbase + v) * HH + h0 + n];
#pragma unroll 4
  for (int k0 = 0; k0 < NPAD; k0 += 32) {
    v16bf a = load_a_bf16(AtBf, NPAD, i0, k0, lane);
    v16bf b = load_b_bf16(qT, NPAD, h0, k0, lane);
    c0 = wmma_bf16(a, b, c0);
  }

  // ---- Phase B: four gate-quadrant tiles, A-frag shared across quadrants ----
  v8f gi, gf, gg, go;
  {
    float b0 = b_ih[0 * HH + h0 + n], b1 = b_ih[1 * HH + h0 + n];
    float b2 = b_ih[2 * HH + h0 + n], b3 = b_ih[3 * HH + h0 + n];
#pragma unroll
    for (int v = 0; v < 8; ++v) { gi[v] = b0; gf[v] = b1; gg[v] = b2; go[v] = b3; }
  }
#pragma unroll
  for (int k0 = 0; k0 < KX; k0 += 32) {            // x-part (K=64, from LDS)
    v16bf a = load_a_bf16(xsh, KX, 0, k0, lane);
    gi = wmma_bf16(a, load_b_bf16(WxT, KX, 0 * HH + h0, k0, lane), gi);
    gf = wmma_bf16(a, load_b_bf16(WxT, KX, 1 * HH + h0, k0, lane), gf);
    gg = wmma_bf16(a, load_b_bf16(WxT, KX, 2 * HH + h0, k0, lane), gg);
    go = wmma_bf16(a, load_b_bf16(WxT, KX, 3 * HH + h0, k0, lane), go);
  }
#pragma unroll
  for (int k0 = 0; k0 < HH; k0 += 32) {            // h-part (K=128)
    v16bf a = load_a_bf16(hbf, HH, i0, k0, lane);
    gi = wmma_bf16(a, load_b_bf16(WhT, HH, 0 * HH + h0, k0, lane), gi);
    gf = wmma_bf16(a, load_b_bf16(WhT, HH, 1 * HH + h0, k0, lane), gf);
    gg = wmma_bf16(a, load_b_bf16(WhT, HH, 2 * HH + h0, k0, lane), gg);
    go = wmma_bf16(a, load_b_bf16(WhT, HH, 3 * HH + h0, k0, lane), go);
  }

  // Fence: all intra-block hbf reads must complete before phase C rewrites it.
  __syncthreads();

  // ---- Phase C: LSTM cell update entirely in registers (ifgo order) ----
#pragma unroll
  for (int v = 0; v < 8; ++v) {
    float cn = fsig_(gf[v]) * c0[v] + fsig_(gi[v]) * ftanh_(gg[v]);
    float hn = fsig_(go[v]) * ftanh_(cn);
    size_t off = (size_t)(i0 + mbase + v) * HH + h0 + n;
    cstate[off] = cn;
    hstate[off] = hn;
    hbf[off]    = (bf16t)hn;
  }
}

// Final projection: out = h @ W_g^T + b_g  (small, f32 VALU)
__global__ void k_final(const float* __restrict__ h, const float* __restrict__ W_g,
                        const float* __restrict__ b_g, float* __restrict__ out) {
  unsigned idx = blockIdx.x * 256u + threadIdx.x;
  if (idx >= (unsigned)NN * OUTD) return;
  int i = idx / OUTD, o = idx % OUTD;
  const float* hr = h + (size_t)i * HH;
  const float* wr = W_g + (size_t)o * HH;
  float s = b_g[o];
#pragma unroll 8
  for (int k = 0; k < HH; ++k) s += hr[k] * wr[k];
  out[idx] = s;
}

// ---------------- Host side ----------------

extern "C" void kernel_launch(void* const* d_in, const int* in_sizes, int n_in,
                              void* d_out, int out_size, void* d_ws, size_t ws_size,
                              hipStream_t stream) {
  const float* x_d  = (const float*)d_in[0];
  const float* x_s  = (const float*)d_in[1];
  const float* A    = (const float*)d_in[2];
  const float* W_ih = (const float*)d_in[3];
  const float* W_hh = (const float*)d_in[4];
  const float* b_ih = (const float*)d_in[5];
  const float* W_q  = (const float*)d_in[6];
  const float* b_q  = (const float*)d_in[7];
  const float* W_g  = (const float*)d_in[8];
  const float* b_g  = (const float*)d_in[9];
  (void)in_sizes; (void)n_in; (void)out_size; (void)ws_size;

  char* ws = (char*)d_ws;
  size_t off = 0;
  auto carve = [&](size_t bytes) {
    void* p = ws + off;
    off = (off + bytes + 255) & ~(size_t)255;
    return p;
  };
  float* hstate = (float*)carve((size_t)NN * HH * 4);
  float* cstate = (float*)carve((size_t)NN * HH * 4);
  bf16t* hbf    = (bf16t*)carve((size_t)NN * HH * 2);
  bf16t* qT     = (bf16t*)carve((size_t)HH * NPAD * 2);
  bf16t* AtBf   = (bf16t*)carve((size_t)NN * NPAD * 2);
  bf16t* WxT    = (bf16t*)carve((size_t)G4 * KX * 2);
  bf16t* WhT    = (bf16t*)carve((size_t)G4 * HH * 2);
  bf16t* WqBf   = (bf16t*)carve((size_t)HH * HH * 2);

  // One-time preprocessing (runs every call; deterministic + idempotent).
  k_setup_A<<<((unsigned)NN * NPAD + 255) / 256, 256, 0, stream>>>(A, AtBf, qT);
  k_setup_W<<<((unsigned)NN * HH + 255) / 256, 256, 0, stream>>>(
      W_ih, W_hh, W_q, WxT, WhT, WqBf, hstate, cstate, hbf);

  // Sequential recurrence: 2 kernels per timestep.
  for (int t = 0; t < TT; ++t) {
    k1_step<<<NN / 16, 256, 0, stream>>>(hbf, WqBf, b_q, qT);
    k2_step<<<NN / 16, 256, 0, stream>>>(t, x_d, x_s, AtBf, qT, WxT, WhT, b_ih,
                                         hstate, cstate, hbf);
  }

  k_final<<<((unsigned)NN * OUTD + 255) / 256, 256, 0, stream>>>(
      hstate, W_g, b_g, (float*)d_out);
}
